// MultiHeadAttention_15547781611871
// MI455X (gfx1250) — matrix operand order repaired and checked by reading, not compiled, not run on hardware
//
#include <hip/hip_runtime.h>
#include <hip/hip_bf16.h>
#include <stdint.h>

// ---------------------------------------------------------------------------
// MultiHeadAttention (B=2, T=2048, D=1024, H=16, DK=64) for gfx1250 / MI455X.
//
// Identity: einsum 'bhtl,bthd->bhtd' sums softmax A over l while v is indexed
// by t, so o == v (softmax rows sum to 1 for any finite mask). Therefore
//   out = perm(x @ Wv) @ Wo^T
// where perm is the raw (b,h,t,dk)->(b,t,d) reshape. Wq/Wk/mask are
// mathematically irrelevant. ~17 GFLOP, ~75 MB traffic -> memory-floor
// ~3.2 us at 23.3 TB/s; bf16 WMMA (f32 accum) keeps compute at that floor.
//
// GEMM engine: 128x128 block tile, 8 waves (wave32), each wave 32x64 via
// 2x4 v_wmma_f32_16x16x32_bf16. Tiles staged with CDNA5 async-to-LDS
// (global_load_async_to_lds_b128, ASYNCcnt), double-buffered LDS, one
// workgroup barrier per K-step.
// ---------------------------------------------------------------------------

typedef __attribute__((ext_vector_type(16))) __bf16 v16bf;
typedef __attribute__((ext_vector_type(8)))  __bf16 bf16x8;
typedef __attribute__((ext_vector_type(8)))  float  v8f;

#define M_ROWS 4096   // B*T
#define K_DIM  1024
#define N_DIM  1024

// ----------------------------- convert kernels -----------------------------

__global__ __launch_bounds__(256) void cvt_copy_bf16(
    const float* __restrict__ src, __bf16* __restrict__ dst, int n) {
  int i = blockIdx.x * blockDim.x + threadIdx.x;
  if (i < n) dst[i] = (__bf16)src[i];
}

// dst[n][k] = src[k][n], 1024x1024, LDS-tiled so both sides stay coalesced
__global__ __launch_bounds__(256) void cvt_transpose_bf16(
    const float* __restrict__ src, __bf16* __restrict__ dst) {
  __shared__ float tile[32][33];
  const int k0 = blockIdx.y * 32, n0 = blockIdx.x * 32;
  const int tx = threadIdx.x & 31, ty = threadIdx.x >> 5;  // ty: 0..7
#pragma unroll
  for (int i = 0; i < 32; i += 8)
    tile[ty + i][tx] = src[(k0 + ty + i) * 1024 + n0 + tx];
  __syncthreads();
#pragma unroll
  for (int i = 0; i < 32; i += 8)
    dst[(n0 + ty + i) * 1024 + k0 + tx] = (__bf16)tile[tx][ty + i];
}

// ------------------------------- WMMA GEMM ---------------------------------

// A-operand element index. GATHER fuses the (b,h,t,dk)->(b,t,d) reshape:
//   perm(V)[row, k] = V[(row/2048)*2048 + (row%128)*16 + k/64,
//                      ((row%2048)/128)*64 + (k%64)]
template <bool GATHER>
__device__ __forceinline__ long a_index(int row, int k) {
  if (GATHER) {
    const int b = row >> 11, tp = row & 2047, h = tp >> 7, r = tp & 127;
    return (long)((b << 11) + (r << 4) + (k >> 6)) * 1024 + (h << 6) + (k & 63);
  }
  return (long)row * K_DIM + k;
}

// CDNA5 async copy: 32 contiguous bytes global -> LDS, no VGPR round-trip.
// INST_OFFSET is applied to BOTH the LDS and the global address (ISA 08 §4.4),
// so the second chunk is the same operands with offset:16. Tracked by ASYNCcnt.
__device__ __forceinline__ void async_copy32(const __bf16* gsrc, __bf16* ldst) {
  const unsigned lds_off = (unsigned)(uintptr_t)ldst;           // addr[31:0] = LDS byte addr
  const unsigned long long gaddr = (unsigned long long)(uintptr_t)gsrc;
  asm volatile(
      "global_load_async_to_lds_b128 %0, %1, off\n\t"
      "global_load_async_to_lds_b128 %0, %1, off offset:16"
      :: "v"(lds_off), "v"(gaddr) : "memory");
}

// C = A(MxK) * B(KxN), B passed pre-transposed as BT[n][k] (row-major in n).
template <bool GATHER, bool OUT_BF16>
__global__ __launch_bounds__(256) void gemm_wmma(
    const __bf16* __restrict__ A, const __bf16* __restrict__ BT,
    float* __restrict__ Cf, __bf16* __restrict__ Cb) {
  __shared__ __bf16 ldsA[2][128 * 32];  // [m][k], row stride 32, double-buffered
  __shared__ __bf16 ldsB[2][128 * 32];  // [n][k], row stride 32, double-buffered

  const int tid  = threadIdx.x;
  const int lane = tid & 31;
  const int wid  = tid >> 5;          // 8 waves (wave32)
  const int wm   = (wid & 3) * 32;    // wave M offset inside block
  const int wn   = (wid >> 2) * 64;   // wave N offset inside block
  const int row0 = blockIdx.y * 128;
  const int n0   = blockIdx.x * 128;

  // staging assignment: each thread owns one (row, 16-elem chunk) per tile
  const int lr = tid >> 1;            // 0..127
  const int lc = (tid & 1) * 16;      // 0 or 16

  v8f acc[2][4] = {};

  // fragment addressing per documented wave32 layouts
  const int fr    = lane & 15;          // M (A) / N (B) row inside 16x16 tile
  const int koff  = (lane >> 4) * 8;    // A: halves = K[koff..+7],[koff+16..+23]
  const int kbase = (lane >> 4) * 16;   // B: same K order as A (uses koff)

  // prologue: stage K-step 0 into buffer 0
  async_copy32(&A[a_index<GATHER>(row0 + lr, lc)],       &ldsA[0][lr * 32 + lc]);
  async_copy32(&BT[(long)(n0 + lr) * K_DIM + lc],        &ldsB[0][lr * 32 + lc]);

  for (int i = 0; i < K_DIM / 32; ++i) {
    const int sel = i & 1;
    // my async loads for buffer `sel` have landed (in-order completion)...
    asm volatile("s_wait_asynccnt 0x0" ::: "memory");
    // ...and after this barrier everyone's have, and all waves are done
    // reading buffer sel^1 (their previous compute precedes this barrier).
    __syncthreads();

    if (i + 1 < K_DIM / 32) {  // overlap: stage next tile during compute
      const int k0n = (i + 1) * 32;
      async_copy32(&A[a_index<GATHER>(row0 + lr, k0n + lc)],
                   &ldsA[sel ^ 1][lr * 32 + lc]);
      async_copy32(&BT[(long)(n0 + lr) * K_DIM + k0n + lc],
                   &ldsB[sel ^ 1][lr * 32 + lc]);
    }

    // ---- build fragments ----
    v16bf afr[2], bfr[4];
#pragma unroll
    for (int mt = 0; mt < 2; ++mt) {
      const __bf16* p = &ldsA[sel][(wm + mt * 16 + fr) * 32 + koff];
      bf16x8 lo = *(const bf16x8*)p;
      bf16x8 hi = *(const bf16x8*)(p + 16);
#pragma unroll
      for (int j = 0; j < 8; ++j) { afr[mt][j] = lo[j]; afr[mt][8 + j] = hi[j]; }
    }
#pragma unroll
    for (int nt = 0; nt < 4; ++nt) {
      const __bf16* p = &ldsB[sel][(wn + nt * 16 + fr) * 32 + koff];
      bf16x8 lo = *(const bf16x8*)p;
      bf16x8 hi = *(const bf16x8*)(p + 16);
#pragma unroll
      for (int j = 0; j < 8; ++j) { bfr[nt][j] = lo[j]; bfr[nt][8 + j] = hi[j]; }
    }

    // ---- 2x4 WMMA tile: D = A x B + C ----
#pragma unroll
    for (int mt = 0; mt < 2; ++mt)
#pragma unroll
      for (int nt = 0; nt < 4; ++nt)
        acc[mt][nt] = __builtin_amdgcn_wmma_f32_16x16x32_bf16(
            false, afr[mt], false, bfr[nt], (short)0, acc[mt][nt], false, false);
  }

  // ---- epilogue: C/D layout = lane(n=lane%16), vgpr j -> m = (lane/16)*8+j
  const int nlo = lane & 15;
  const int mhi = (lane >> 4) * 8;
#pragma unroll
  for (int mt = 0; mt < 2; ++mt) {
#pragma unroll
    for (int nt = 0; nt < 4; ++nt) {
      const int ncol = n0 + wn + nt * 16 + nlo;
#pragma unroll
      for (int j = 0; j < 8; ++j) {
        const int mrow = row0 + wm + mt * 16 + mhi + j;
        if (OUT_BF16) Cb[(long)mrow * N_DIM + ncol] = (__bf16)acc[mt][nt][j];
        else          Cf[(long)mrow * N_DIM + ncol] = acc[mt][nt][j];
      }
    }
  }
}

// ------------------------------- launcher ----------------------------------

extern "C" void kernel_launch(void* const* d_in, const int* in_sizes, int n_in,
                              void* d_out, int out_size, void* d_ws, size_t ws_size,
                              hipStream_t stream) {
  // inputs: 0=x, 1=mask(unused), 2=Wq(unused), 3=Wk(unused), 4=Wv, 5=Wo
  const float* x  = (const float*)d_in[0];
  const float* Wv = (const float*)d_in[4];
  const float* Wo = (const float*)d_in[5];
  float* out = (float*)d_out;

  char* ws = (char*)d_ws;
  __bf16* Xbf  = (__bf16*)(ws);                          // 4096*1024*2 = 8 MB
  __bf16* WvT  = (__bf16*)(ws + (8u  << 20));            // 2 MB
  __bf16* Wobf = (__bf16*)(ws + (10u << 20));            // 2 MB
  __bf16* Vbf  = (__bf16*)(ws + (12u << 20));            // 8 MB

  cvt_copy_bf16<<<(M_ROWS * K_DIM) / 256, 256, 0, stream>>>(x, Xbf, M_ROWS * K_DIM);
  cvt_transpose_bf16<<<dim3(32, 32), 256, 0, stream>>>(Wv, WvT);
  cvt_copy_bf16<<<(K_DIM * N_DIM) / 256, 256, 0, stream>>>(Wo, Wobf, K_DIM * N_DIM);

  // GEMM1: Vbf = Xbf @ Wv  (store bf16)
  gemm_wmma<false, true><<<dim3(N_DIM / 128, M_ROWS / 128), 256, 0, stream>>>(
      Xbf, WvT, nullptr, Vbf);
  // GEMM2: out = perm(Vbf) @ Wo^T  (gather fused into A load, store f32)
  gemm_wmma<true, false><<<dim3(N_DIM / 128, M_ROWS / 128), 256, 0, stream>>>(
      Vbf, Wobf, out, nullptr);
}
